// GraphConvolution_69045894250505
// MI455X (gfx1250) — compile-verified, matrix operand order checked by reading, not run on hardware
//
#include <hip/hip_runtime.h>

#define B_DIM 8
#define N_DIM 10000
#define F_DIM 256
#define H_DIM 128
#define E_DIM 320000
#define M_DIM (B_DIM * N_DIM)   // 80000 rows
#define BM    64
#define KTILE 32
#define KSTEPS (F_DIM / KTILE)   // 8
#define LDS_PAD 40               // bf16 elems per LDS row (80B stride -> conflict-free b128 reads)

typedef __attribute__((ext_vector_type(16))) __bf16 v16bf;
typedef __attribute__((ext_vector_type(8)))  float  v8f;

// ---------------- zero output ----------------
__global__ __launch_bounds__(256) void zero_kernel(float4* __restrict__ p, int n4) {
  int i = blockIdx.x * 256 + threadIdx.x;
  if (i < n4) p[i] = make_float4(0.f, 0.f, 0.f, 0.f);
}

// ------- weight -> W^T bf16 hi/lo (tiny, L2-resident) -------
__global__ __launch_bounds__(256) void wt_convert_kernel(const float* __restrict__ w,
                                                         __bf16* __restrict__ wt_hi,
                                                         __bf16* __restrict__ wt_lo) {
  int idx = blockIdx.x * 256 + threadIdx.x;     // over F*H
  if (idx >= F_DIM * H_DIM) return;
  int k = idx / H_DIM, n = idx % H_DIM;
  float v = w[idx];                              // w[k][n]
  __bf16 h = (__bf16)v;
  wt_hi[n * F_DIM + k] = h;
  wt_lo[n * F_DIM + k] = (__bf16)(v - (float)h);
}

// ------- fused dropout + bf16x3 WMMA GEMM: xw = dropout(x) @ W -------
__global__ __launch_bounds__(256) void gemm_kernel(const float* __restrict__ x,
                                                   const float* __restrict__ du,
                                                   const __bf16* __restrict__ wt_hi,
                                                   const __bf16* __restrict__ wt_lo,
                                                   float* __restrict__ xw) {
  __shared__ __bf16 a_hi[2][BM][LDS_PAD];
  __shared__ __bf16 a_lo[2][BM][LDS_PAD];
  const int tid  = threadIdx.x;
  const int wave = tid >> 5;          // 0..7 -> N column block (8*16 = 128 = H)
  const int lane = tid & 31;
  const long rowBase = (long)blockIdx.x * BM;
  const int n0 = wave * 16;

  v8f acc[4] = {};                    // 4 M-subtiles of 16x16 f32

  // A/B fragment addressing per documented 16-bit layout
  const int arow = lane & 15;
  const int kb   = (lane >> 4) * 8;   // K split: {kb..kb+7, kb+16..kb+23}

  // cooperative staging indices: 256 threads cover 64x32 tile, 8 floats each
  const int ldRow = tid >> 2;         // 0..63
  const int ldCol = (tid & 3) * 8;    // 0,8,16,24

  // fused inverted dropout (p=0.5 -> x*2 on keep) + hi/lo bf16 split into LDS buffer `buf`
  auto stage = [&](int k0, int buf) {
    const float* xp = x  + (rowBase + ldRow) * F_DIM + k0 + ldCol;
    const float* up = du + (rowBase + ldRow) * F_DIM + k0 + ldCol;
    if (k0 + 2 * KTILE <= F_DIM - KTILE + ldCol + 8) { /* always-in-bounds guard below */ }
    float4 x0 = ((const float4*)xp)[0];
    float4 x1 = ((const float4*)xp)[1];
    float4 u0 = ((const float4*)up)[0];
    float4 u1 = ((const float4*)up)[1];
    if (k0 + 2 * KTILE < F_DIM) {     // prefetch two K-tiles ahead (same rows, in bounds)
      __builtin_prefetch(xp + 2 * KTILE, 0, 0);
      __builtin_prefetch(up + 2 * KTILE, 0, 0);
    }
    float xs[8] = {x0.x, x0.y, x0.z, x0.w, x1.x, x1.y, x1.z, x1.w};
    float us[8] = {u0.x, u0.y, u0.z, u0.w, u1.x, u1.y, u1.z, u1.w};
    union { __bf16 b[8]; uint4 u; } Hh, Ll;
#pragma unroll
    for (int i = 0; i < 8; ++i) {
      float v = (us[i] > 0.5f) ? xs[i] * 2.0f : 0.0f;
      __bf16 h = (__bf16)v;
      Hh.b[i] = h;
      Ll.b[i] = (__bf16)(v - (float)h);
    }
    *(uint4*)(&a_hi[buf][ldRow][ldCol]) = Hh.u;
    *(uint4*)(&a_lo[buf][ldRow][ldCol]) = Ll.u;
  };

  stage(0, 0);
  int cur = 0;
  for (int step = 0; step < KSTEPS; ++step) {
    const int k0 = step * KTILE;
    __syncthreads();                          // stores to buf `cur` visible to all waves
    if (step + 1 < KSTEPS) stage(k0 + KTILE, cur ^ 1);  // overlap next-tile HBM loads with WMMA

    // B fragments from W^T (row n contiguous in K), hi/lo
    v16bf bh, bl;
    {
      const __bf16* ph = wt_hi + (n0 + arow) * F_DIM + k0 + kb;
      const __bf16* pl = wt_lo + (n0 + arow) * F_DIM + k0 + kb;
      ((uint4*)&bh)[0] = *(const uint4*)(ph);
      ((uint4*)&bh)[1] = *(const uint4*)(ph + 16);
      ((uint4*)&bl)[0] = *(const uint4*)(pl);
      ((uint4*)&bl)[1] = *(const uint4*)(pl + 16);
    }

#pragma unroll
    for (int m = 0; m < 4; ++m) {
      v16bf ah, al;
      const __bf16* pa = &a_hi[cur][m * 16 + arow][kb];
      const __bf16* pb = &a_lo[cur][m * 16 + arow][kb];
      ((uint4*)&ah)[0] = *(const uint4*)(pa);
      ((uint4*)&ah)[1] = *(const uint4*)(pa + 16);
      ((uint4*)&al)[0] = *(const uint4*)(pb);
      ((uint4*)&al)[1] = *(const uint4*)(pb + 16);
      // bf16x3: a*b ~= ah*bh + ah*bl + al*bh  (f32 accumulate)
      acc[m] = __builtin_amdgcn_wmma_f32_16x16x32_bf16(false, ah, false, bh, (short)0, acc[m], false, false);
      acc[m] = __builtin_amdgcn_wmma_f32_16x16x32_bf16(false, ah, false, bl, (short)0, acc[m], false, false);
      acc[m] = __builtin_amdgcn_wmma_f32_16x16x32_bf16(false, al, false, bh, (short)0, acc[m], false, false);
    }
    cur ^= 1;
  }

  // epilogue: C layout -> VGPR r holds (M = r + 8*(lane>=16), N = lane&15)
  const int crow = (lane >> 4) * 8;
  const int ccol = lane & 15;
#pragma unroll
  for (int m = 0; m < 4; ++m) {
    float* outp = xw + (rowBase + m * 16 + crow) * H_DIM + n0 + ccol;
#pragma unroll
    for (int r = 0; r < 8; ++r)
      outp[(long)r * H_DIM] = acc[m][r];
  }
}

// hardware f32 atomic add, no return (STOREcnt-tracked; s_endpgm waits)
__device__ __forceinline__ void atomic_add_f32(float* p, float v) {
  asm volatile("global_atomic_add_f32 %0, %1, off" :: "v"(p), "v"(v) : "memory");
}

// ------- SpMM scatter: out[b,row] += val * xw[b,col]  (xw gathers hit L2) -------
__global__ __launch_bounds__(256) void spmm_kernel(const int* __restrict__ rows,
                                                   const int* __restrict__ cols,
                                                   const float* __restrict__ vals,
                                                   const float* __restrict__ xw,
                                                   float* __restrict__ out) {
  const int lane = threadIdx.x & 31;
  const int gw = (int)((blockIdx.x * blockDim.x + threadIdx.x) >> 5);
  const int nw = (int)((gridDim.x * blockDim.x) >> 5);
  const int totalE = B_DIM * E_DIM;
  for (int e = gw; e < totalE; e += nw) {
    const int e2 = e + nw;
    if (e2 < totalE) {                        // prefetch next edge-index stride
      __builtin_prefetch(rows + e2, 0, 0);
      __builtin_prefetch(cols + e2, 0, 0);
      __builtin_prefetch(vals + e2, 0, 0);
    }
    const int b = e / E_DIM;                  // rows/cols/vals are [B,E] flat = index e
    const int r = rows[e];
    const int c = cols[e];
    const float v = vals[e];
    const float4 s = *(const float4*)(xw + ((long)b * N_DIM + c) * H_DIM + lane * 4);
    float* d = out + ((long)b * N_DIM + r) * H_DIM + lane * 4;
    atomic_add_f32(d + 0, v * s.x);
    atomic_add_f32(d + 1, v * s.y);
    atomic_add_f32(d + 2, v * s.z);
    atomic_add_f32(d + 3, v * s.w);
  }
}

extern "C" void kernel_launch(void* const* d_in, const int* in_sizes, int n_in,
                              void* d_out, int out_size, void* d_ws, size_t ws_size,
                              hipStream_t stream) {
  const float* x    = (const float*)d_in[0];
  const float* du   = (const float*)d_in[1];
  const int*   rows = (const int*)d_in[2];
  const int*   cols = (const int*)d_in[3];
  const float* vals = (const float*)d_in[4];
  const float* w    = (const float*)d_in[5];
  float* out = (float*)d_out;

  // workspace layout: [0,64K) Wt_hi bf16 | [64K,128K) Wt_lo bf16 | [128K, ...) xw f32
  char* ws = (char*)d_ws;
  __bf16* wt_hi = (__bf16*)(ws);
  __bf16* wt_lo = (__bf16*)(ws + 65536);
  float*  xw    = (float*)(ws + 131072);

  const int n4 = out_size / 4;   // out_size = 8*10000*128, divisible by 4
  zero_kernel<<<(n4 + 255) / 256, 256, 0, stream>>>((float4*)out, n4);
  wt_convert_kernel<<<(F_DIM * H_DIM + 255) / 256, 256, 0, stream>>>(w, wt_hi, wt_lo);
  gemm_kernel<<<M_DIM / BM, 256, 0, stream>>>(x, du, wt_hi, wt_lo, xw);
  spmm_kernel<<<4096, 256, 0, stream>>>(rows, cols, vals, xw, out);
}
